// TransformerBlock_66133906424111
// MI455X (gfx1250) — compile-verified, matrix operand order checked by reading, not run on hardware
//
#include <hip/hip_runtime.h>

// ---------------------------------------------------------------------------
// Types for CDNA5 WMMA (wave32): v_wmma_f32_16x16x32_bf16
// ---------------------------------------------------------------------------
typedef __bf16 bf16_t;
typedef bf16_t v16bf __attribute__((ext_vector_type(16)));
typedef bf16_t v8bf  __attribute__((ext_vector_type(8)));
typedef float  v8f   __attribute__((ext_vector_type(8)));

union FragU { v16bf v; v8bf h[2]; };
union PackU { v8bf v; bf16_t e[8]; };

// ---------------------------------------------------------------------------
// fp32 -> bf16 elementwise
// ---------------------------------------------------------------------------
__global__ __launch_bounds__(256)
void cvt_f32_to_bf16(const float* __restrict__ in, bf16_t* __restrict__ out, long long n) {
  long long base = (long long)blockIdx.x * 1024 + threadIdx.x;
#pragma unroll
  for (int i = 0; i < 4; ++i) {
    long long j = base + (long long)i * 256;
    if (j < n) out[j] = (bf16_t)in[j];
  }
}

// fp32 W[K=1024][N=1024] -> bf16 Wt[N][K]  (so GEMM consumes B as N x K row-major)
__global__ __launch_bounds__(256)
void cvt_weight_t(const float* __restrict__ W, bf16_t* __restrict__ Wt) {
  const int n = blockIdx.x;              // output row (original column)
  for (int k = threadIdx.x; k < 1024; k += 256)
    Wt[(long long)n * 1024 + k] = (bf16_t)W[(long long)k * 1024 + n];
}

// ---------------------------------------------------------------------------
// bf16 WMMA GEMM:  C[M x N] = A[M x K] * B'[N x K]^T   (fp32 accumulate)
//   block tile 256x128, 8 waves, wave tile 64x64 (16 WMMA / 16 ds_load_b128
//   per K-step), K-step 32, LDS double buffer with 80B padded rows.
//   epilogue: optional bias (fp32), leaky-relu(0.2), transposed store (packed)
//   M % 256 == 0, N % 128 == 0, K % 32 == 0 (guaranteed by the launcher)
// ---------------------------------------------------------------------------
template<bool BIAS, bool LEAKY, bool STORE_T>
__global__ __launch_bounds__(256)
void gemm_bf16_wmma(const bf16_t* __restrict__ A, long long sA,
                    const bf16_t* __restrict__ Bm, long long sB,
                    bf16_t* __restrict__ C, long long sC,
                    const float* __restrict__ bias,
                    int M, int N, int Kd, int lda, int ldb, int ldc) {
  (void)M; (void)N;
  constexpr int LR = 40;  // 32 k-elems + 8 pad -> 80B row stride (16B aligned)
  __shared__ __attribute__((aligned(16))) bf16_t As[2][256 * LR];  // 40 KB
  __shared__ __attribute__((aligned(16))) bf16_t Bs[2][128 * LR];  // 20 KB

  const int bz = blockIdx.z;
  const bf16_t* Ab = A  + (long long)bz * sA;
  const bf16_t* Bb = Bm + (long long)bz * sB;
  bf16_t*       Cb = C  + (long long)bz * sC;

  const int m0 = blockIdx.y * 256;
  const int n0 = blockIdx.x * 128;
  const int tid  = threadIdx.x;
  const int lane = tid & 31;
  const int wave = tid >> 5;
  const int wm = (wave & 3) * 64;   // wave M offset within block tile (4 waves)
  const int wn = (wave >> 2) * 64;  // wave N offset within block tile (2 waves)
  const int lm = lane & 15;         // row-within-fragment
  const int lh = lane >> 4;         // half-wave selector (k-interleave)

  v8f acc[4][4] = {};

  // Global tile loaders: 256 threads, 16B chunks.
  //   A tile 256x32 bf16 = 1024 chunks -> 4 per thread
  //   B tile 128x32 bf16 =  512 chunks -> 2 per thread
  const int r0 = tid >> 2;              // rows 0..63 (passes add +64p)
  const int c0 = (tid & 3) * 8;         // k-element offset (8 bf16 = 16B)
  const bf16_t* Ag[4];
  const bf16_t* Bg[2];
#pragma unroll
  for (int p = 0; p < 4; ++p) Ag[p] = Ab + (long long)(m0 + r0 + 64 * p) * lda + c0;
#pragma unroll
  for (int p = 0; p < 2; ++p) Bg[p] = Bb + (long long)(n0 + r0 + 64 * p) * ldb + c0;

  const int nk = Kd >> 5;

  auto load_tile = [&](int kt, int buf) {
    const long long ko = (long long)kt * 32;
    v8bf a[4], b[2];
#pragma unroll
    for (int p = 0; p < 4; ++p) a[p] = *(const v8bf*)(Ag[p] + ko);
#pragma unroll
    for (int p = 0; p < 2; ++p) b[p] = *(const v8bf*)(Bg[p] + ko);
#pragma unroll
    for (int p = 0; p < 4; ++p) *(v8bf*)&As[buf][(r0 + 64 * p) * LR + c0] = a[p];
#pragma unroll
    for (int p = 0; p < 2; ++p) *(v8bf*)&Bs[buf][(r0 + 64 * p) * LR + c0] = b[p];
  };

  load_tile(0, 0);
  __syncthreads();

  for (int kt = 0; kt < nk; ++kt) {
    const int buf = kt & 1;
    if (kt + 1 < nk) load_tile(kt + 1, buf ^ 1);
    if (kt + 2 < nk) {  // L2 prefetch two K-tiles ahead (global_prefetch_b8)
      __builtin_prefetch(Ag[0] + (long long)(kt + 2) * 32, 0, 0);
      __builtin_prefetch(Bg[0] + (long long)(kt + 2) * 32, 0, 0);
    }

    // Fragment loads from LDS per the 16-bit A/B VGPR layout:
    //   lanes 0-15: k = 0..7 (VGPR0-3) and 16..23 (VGPR4-7)
    //   lanes 16-31: k = 8..15 and 24..31
    v16bf af[4], bfr[4];
#pragma unroll
    for (int mi = 0; mi < 4; ++mi) {
      const bf16_t* p = &As[buf][(wm + mi * 16 + lm) * LR];
      FragU u;
      u.h[0] = *(const v8bf*)(p + lh * 8);
      u.h[1] = *(const v8bf*)(p + 16 + lh * 8);
      af[mi] = u.v;
    }
#pragma unroll
    for (int ni = 0; ni < 4; ++ni) {
      const bf16_t* p = &Bs[buf][(wn + ni * 16 + lm) * LR];
      FragU u;
      u.h[0] = *(const v8bf*)(p + lh * 8);
      u.h[1] = *(const v8bf*)(p + 16 + lh * 8);
      bfr[ni] = u.v;
    }
#pragma unroll
    for (int mi = 0; mi < 4; ++mi)
#pragma unroll
      for (int ni = 0; ni < 4; ++ni)
        acc[mi][ni] = __builtin_amdgcn_wmma_f32_16x16x32_bf16(
            false, af[mi], false, bfr[ni], (short)0, acc[mi][ni], false, false);

    __syncthreads();
  }

  // Epilogue. C layout: VGPR r holds row (lh*8 + r), lane%16 = column.
#pragma unroll
  for (int mi = 0; mi < 4; ++mi) {
#pragma unroll
    for (int ni = 0; ni < 4; ++ni) {
      const int col  = n0 + wn + ni * 16 + lm;
      const int row0 = m0 + wm + mi * 16 + lh * 8;
      float bv = 0.f;
      if (BIAS) bv = bias[col];
      if (STORE_T) {
        // Transposed store: lane owns 8 contiguous elements along the fast dim.
        PackU pk;
#pragma unroll
        for (int r = 0; r < 8; ++r) {
          float v = acc[mi][ni][r] + bv;
          if (LEAKY) v = (v > 0.f) ? v : 0.2f * v;
          pk.e[r] = (bf16_t)v;
        }
        *(v8bf*)&Cb[(long long)col * ldc + row0] = pk.v;
      } else {
#pragma unroll
        for (int r = 0; r < 8; ++r) {
          float v = acc[mi][ni][r] + bv;
          if (LEAKY) v = (v > 0.f) ? v : 0.2f * v;
          Cb[(long long)(row0 + r) * ldc + col] = (bf16_t)v;
        }
      }
    }
  }
}

// ---------------------------------------------------------------------------
// Row softmax on bf16 scores, in place. cols == 2048. scale = 1/sqrt(Lk).
// Causal: elements j > row masked to -inf.
// ---------------------------------------------------------------------------
template<bool CAUSAL>
__global__ __launch_bounds__(256)
void softmax_kernel(bf16_t* __restrict__ P, int cols, float scale) {
  __shared__ float red[256];
  const int row = blockIdx.x;
  const int b   = blockIdx.y;
  bf16_t* p = P + ((long long)b * gridDim.x + row) * cols;
  const int tid = threadIdx.x;

  float sv[8];
  float mx = -3.0e38f;
#pragma unroll
  for (int i = 0; i < 8; ++i) {
    int j = tid + (i << 8);
    float x = (float)p[j] * scale;
    bool valid = (!CAUSAL) || (j <= row);
    sv[i] = valid ? x : -3.0e38f;
    mx = fmaxf(mx, sv[i]);
  }
  red[tid] = mx; __syncthreads();
  for (int s = 128; s > 0; s >>= 1) { if (tid < s) red[tid] = fmaxf(red[tid], red[tid + s]); __syncthreads(); }
  const float m = red[0];
  __syncthreads();

  float sum = 0.f, ex[8];
#pragma unroll
  for (int i = 0; i < 8; ++i) {
    ex[i] = (sv[i] <= -1.0e38f) ? 0.f : __expf(sv[i] - m);
    sum += ex[i];
  }
  red[tid] = sum; __syncthreads();
  for (int s = 128; s > 0; s >>= 1) { if (tid < s) red[tid] += red[tid + s]; __syncthreads(); }
  const float inv = 1.f / red[0];
#pragma unroll
  for (int i = 0; i < 8; ++i) {
    int j = tid + (i << 8);
    p[j] = (bf16_t)(ex[i] * inv);
  }
}

// ---------------------------------------------------------------------------
// out = LayerNorm(x + r) as bf16. Row length 1024, one row per block.
// ---------------------------------------------------------------------------
__global__ __launch_bounds__(256)
void add_layernorm_kernel(const bf16_t* __restrict__ x, const bf16_t* __restrict__ r,
                          const float* __restrict__ gamma, const float* __restrict__ beta,
                          bf16_t* __restrict__ out) {
  __shared__ float red[256];
  const long long base = (long long)blockIdx.x * 1024;
  const int tid = threadIdx.x;
  float v[4], s = 0.f;
#pragma unroll
  for (int i = 0; i < 4; ++i) {
    int j = tid + (i << 8);
    v[i] = (float)x[base + j] + (float)r[base + j];
    s += v[i];
  }
  red[tid] = s; __syncthreads();
  for (int t = 128; t > 0; t >>= 1) { if (tid < t) red[tid] += red[tid + t]; __syncthreads(); }
  const float mu = red[0] * (1.f / 1024.f);
  __syncthreads();
  float q = 0.f;
#pragma unroll
  for (int i = 0; i < 4; ++i) { float d = v[i] - mu; q += d * d; }
  red[tid] = q; __syncthreads();
  for (int t = 128; t > 0; t >>= 1) { if (tid < t) red[tid] += red[tid + t]; __syncthreads(); }
  const float rs = rsqrtf(red[0] * (1.f / 1024.f) + 1e-3f);
#pragma unroll
  for (int i = 0; i < 4; ++i) {
    int j = tid + (i << 8);
    out[base + j] = (bf16_t)((v[i] - mu) * rs * gamma[j] + beta[j]);
  }
}

// out = relu(LayerNorm(ff + res)) as fp32 (final output)
__global__ __launch_bounds__(256)
void final_ln_relu_kernel(const bf16_t* __restrict__ x, const bf16_t* __restrict__ r,
                          const float* __restrict__ gamma, const float* __restrict__ beta,
                          float* __restrict__ out) {
  __shared__ float red[256];
  const long long base = (long long)blockIdx.x * 1024;
  const int tid = threadIdx.x;
  float v[4], s = 0.f;
#pragma unroll
  for (int i = 0; i < 4; ++i) {
    int j = tid + (i << 8);
    v[i] = (float)x[base + j] + (float)r[base + j];
    s += v[i];
  }
  red[tid] = s; __syncthreads();
  for (int t = 128; t > 0; t >>= 1) { if (tid < t) red[tid] += red[tid + t]; __syncthreads(); }
  const float mu = red[0] * (1.f / 1024.f);
  __syncthreads();
  float q = 0.f;
#pragma unroll
  for (int i = 0; i < 4; ++i) { float d = v[i] - mu; q += d * d; }
  red[tid] = q; __syncthreads();
  for (int t = 128; t > 0; t >>= 1) { if (tid < t) red[tid] += red[tid + t]; __syncthreads(); }
  const float rs = rsqrtf(red[0] * (1.f / 1024.f) + 1e-3f);
#pragma unroll
  for (int i = 0; i < 4; ++i) {
    int j = tid + (i << 8);
    float y = (v[i] - mu) * rs * gamma[j] + beta[j];
    out[base + j] = fmaxf(y, 0.f);
  }
}

// ---------------------------------------------------------------------------
// Orchestration
// ---------------------------------------------------------------------------
extern "C" void kernel_launch(void* const* d_in, const int* in_sizes, int n_in,
                              void* d_out, int out_size, void* d_ws, size_t ws_size,
                              hipStream_t stream) {
  (void)in_sizes; (void)n_in; (void)out_size; (void)ws_size;

  const float* inputs = (const float*)d_in[0];
  const float* ctxp   = (const float*)d_in[1];
  const float* Wk1 = (const float*)d_in[2];
  const float* Wq1 = (const float*)d_in[3];
  const float* Wv1 = (const float*)d_in[4];
  const float* Wk2 = (const float*)d_in[5];
  const float* Wq2 = (const float*)d_in[6];
  const float* Wv2 = (const float*)d_in[7];
  const float* W1  = (const float*)d_in[8];
  const float* b1  = (const float*)d_in[9];
  const float* W2  = (const float*)d_in[10];
  const float* b2  = (const float*)d_in[11];
  const float* gamma = (const float*)d_in[12];
  const float* beta  = (const float*)d_in[13];
  float* out = (float*)d_out;

  const int Bn = 8, S = 2048, D = 1024;
  const long long MS  = (long long)Bn * S;   // 16384 flat tokens
  const long long MSD = MS * D;              // 16,777,216
  const long long SD  = (long long)S * D;
  const long long SS  = (long long)S * S;
  const long long D2  = (long long)D * D;
  const float SCALE = 0.02209708691f;        // 1/sqrt(2048)

  char* ws = (char*)d_ws;
  size_t off = 0;
  auto alloc = [&](size_t bytes) -> void* {
    void* p = (void*)(ws + off);
    off += (bytes + 255) & ~(size_t)255;
    return p;
  };

  bf16_t* Xb = (bf16_t*)alloc((size_t)MSD * 2);        // inputs bf16
  bf16_t* Cx = (bf16_t*)alloc((size_t)MSD * 2);        // context bf16
  bf16_t* Wt[8];
  for (int i = 0; i < 8; ++i) Wt[i] = (bf16_t*)alloc((size_t)D2 * 2);  // W^T bf16
  bf16_t* Qb = (bf16_t*)alloc((size_t)MSD * 2);        // Q / Q2 / ff
  bf16_t* Kb = (bf16_t*)alloc((size_t)MSD * 2);        // K / K2 / h
  bf16_t* Vt = (bf16_t*)alloc((size_t)MSD * 2);        // V^T / V2^T
  bf16_t* Pb = (bf16_t*)alloc((size_t)Bn * SS * 2);    // scores / probs
  bf16_t* Ab = (bf16_t*)alloc((size_t)MSD * 2);        // att / att2
  bf16_t* N1 = (bf16_t*)alloc((size_t)MSD * 2);        // norm1
  bf16_t* N2 = (bf16_t*)alloc((size_t)MSD * 2);        // norm2

  dim3 blk(256, 1, 1);

  // 0) precision conversion
  cvt_f32_to_bf16<<<dim3(16384), blk, 0, stream>>>(inputs, Xb, MSD);
  cvt_f32_to_bf16<<<dim3(16384), blk, 0, stream>>>(ctxp,   Cx, MSD);
  const float* Wsrc[8] = {Wk1, Wq1, Wv1, Wk2, Wq2, Wv2, W1, W2};
  for (int i = 0; i < 8; ++i)
    cvt_weight_t<<<dim3(1024), blk, 0, stream>>>(Wsrc[i], Wt[i]);

  // 1) self-attention projections (flat M = 16384; grid = {N/128, M/256, z})
  gemm_bf16_wmma<false,false,false><<<dim3(8,64,1), blk, 0, stream>>>(
      Xb,0, Wt[1],0, Qb,0, nullptr, (int)MS,D,D, D,D,D);                  // Q
  gemm_bf16_wmma<false,false,false><<<dim3(8,64,1), blk, 0, stream>>>(
      Xb,0, Wt[0],0, Kb,0, nullptr, (int)MS,D,D, D,D,D);                  // K
  gemm_bf16_wmma<false,false,true><<<dim3(8,8,8), blk, 0, stream>>>(
      Xb,SD, Wt[2],0, Vt,SD, nullptr, S,D,D, D,D,S);                      // V^T (per batch)

  // 2) scores = Q K^T, causal softmax, att = P V
  gemm_bf16_wmma<false,false,false><<<dim3(16,8,8), blk, 0, stream>>>(
      Qb,SD, Kb,SD, Pb,SS, nullptr, S,S,D, D,D,S);
  softmax_kernel<true><<<dim3(S,8), blk, 0, stream>>>(Pb, S, SCALE);
  gemm_bf16_wmma<false,false,false><<<dim3(8,8,8), blk, 0, stream>>>(
      Pb,SS, Vt,SD, Ab,SD, nullptr, S,D,S, S,S,D);

  // 3) norm1 = LN(att + inputs)
  add_layernorm_kernel<<<dim3(16384), blk, 0, stream>>>(Ab, Xb, gamma, beta, N1);

  // 4) cross-attention: K2,V2 from context, Q2 from att (pre-norm)
  gemm_bf16_wmma<false,false,false><<<dim3(8,64,1), blk, 0, stream>>>(
      Cx,0, Wt[3],0, Kb,0, nullptr, (int)MS,D,D, D,D,D);                  // K2
  gemm_bf16_wmma<false,false,true><<<dim3(8,8,8), blk, 0, stream>>>(
      Cx,SD, Wt[5],0, Vt,SD, nullptr, S,D,D, D,D,S);                      // V2^T
  gemm_bf16_wmma<false,false,false><<<dim3(8,64,1), blk, 0, stream>>>(
      Ab,0, Wt[4],0, Qb,0, nullptr, (int)MS,D,D, D,D,D);                  // Q2
  gemm_bf16_wmma<false,false,false><<<dim3(16,8,8), blk, 0, stream>>>(
      Qb,SD, Kb,SD, Pb,SS, nullptr, S,S,D, D,D,S);
  softmax_kernel<false><<<dim3(S,8), blk, 0, stream>>>(Pb, S, SCALE);
  gemm_bf16_wmma<false,false,false><<<dim3(8,8,8), blk, 0, stream>>>(
      Pb,SS, Vt,SD, Ab,SD, nullptr, S,D,S, S,S,D);                        // att2

  // 5) norm2 = LN(att2 + norm1)
  add_layernorm_kernel<<<dim3(16384), blk, 0, stream>>>(Ab, N1, gamma, beta, N2);

  // 6) FFN: h = leaky_relu(norm2 W1 + b1); ff = h W2 + b2
  gemm_bf16_wmma<true,true,false><<<dim3(8,64,1), blk, 0, stream>>>(
      N2,0, Wt[6],0, Kb,0, b1, (int)MS,D,D, D,D,D);
  gemm_bf16_wmma<true,false,false><<<dim3(8,64,1), blk, 0, stream>>>(
      Kb,0, Wt[7],0, Qb,0, b2, (int)MS,D,D, D,D,D);

  // 7) out = relu(LN(ff + norm2)) as fp32
  final_ln_relu_kernel<<<dim3(16384), blk, 0, stream>>>(Qb, N2, gamma, beta, out);
}
